// Pattention_19670950216252
// MI455X (gfx1250) — matrix operand order repaired and checked by reading, not compiled, not run on hardware
//
#include <hip/hip_runtime.h>

// Fused Pattention for MI455X (gfx1250, wave32, WMMA + TDM).
// out[b,t,:] = normalize(gelu(inputs[b,t,:] @ K^T)) * sqrt(P) @ V
// Single pass over P with post-hoc row scaling; f16 WMMA with f32 accumulate.
// K/V chunk staging uses the Tensor Data Mover (tensor_load_to_lds) when the
// builtin is available; falls back to manual vector copies otherwise.

typedef __attribute__((ext_vector_type(16))) _Float16 v16h;
typedef __attribute__((ext_vector_type(8)))  float    v8f;

union FragU { v16h h; unsigned int u[8]; };

constexpr int Bb = 4;
constexpr int Tt = 4096;
constexpr int Dd = 512;
constexpr int Pp = 4096;
constexpr int BM = 64;         // rows of flattened (B*T) per block
constexpr int PC = 64;         // P chunk per iteration
constexpr int NTHREADS = 512;  // 16 waves

#ifndef __has_builtin
#define __has_builtin(x) 0
#endif
#if __has_builtin(__builtin_amdgcn_tensor_load_to_lds) && \
    __has_builtin(__builtin_amdgcn_s_wait_tensorcnt)
#define USE_TDM 1
#endif

#ifdef USE_TDM
typedef __attribute__((ext_vector_type(4))) unsigned int u32x4;
typedef __attribute__((ext_vector_type(4))) int          i32x4;
typedef __attribute__((ext_vector_type(8))) int          i32x8;

// Issue one TDM 2D tile load (global f16 -> LDS, rows packed contiguously).
//   tile_w   : elements per row (data_size = 2 bytes)
//   tile_h   : number of rows
//   stride   : global row stride in elements
// D# layout per CDNA5 ISA.
__device__ __forceinline__ void tdm_load_tile_2d(const void* gptr, void* lptr,
                                                 unsigned tile_w, unsigned tile_h,
                                                 unsigned stride_elems) {
  unsigned long long ga = (unsigned long long)gptr;
  unsigned la = (unsigned)(unsigned long long)lptr;  // LDS byte offset (low 32)
  u32x4 g0;
  g0[0] = 1u;                                          // count=1, user mode
  g0[1] = la;                                          // lds_addr [63:32]
  g0[2] = (unsigned)(ga & 0xFFFFFFFFu);                // global_addr [95:64]
  g0[3] = (unsigned)((ga >> 32) & 0x01FFFFFFu)         // global_addr [120:96]
        | (2u << 30);                                  // type=2 ("image")
  i32x8 g1;
  g1[0] = (int)(1u << 16);                             // data_size=1 -> 2 bytes
  g1[1] = (int)((tile_w & 0xFFFFu) << 16);             // tensor_dim0 lo16
  g1[2] = (int)(((tile_w >> 16) & 0xFFFFu)             // tensor_dim0 hi16
        | ((tile_h & 0xFFFFu) << 16));                 // tensor_dim1 lo16
  g1[3] = (int)(((tile_h >> 16) & 0xFFFFu)             // tensor_dim1 hi16
        | ((tile_w & 0xFFFFu) << 16));                 // tile_dim0
  g1[4] = (int)(tile_h & 0xFFFFu);                     // tile_dim1 (tile_dim2=0)
  g1[5] = (int)stride_elems;                           // tensor_dim0_stride lo32
  g1[6] = 0;                                           // stride hi / dim1_stride lo
  g1[7] = 0;
  i32x4 z4 = {0, 0, 0, 0};
#if __clang_major__ >= 23
  i32x8 z8 = {0, 0, 0, 0, 0, 0, 0, 0};
  __builtin_amdgcn_tensor_load_to_lds(g0, g1, z4, z4, z8, 0);
#else
  __builtin_amdgcn_tensor_load_to_lds(g0, g1, z4, z4, 0);
#endif
}
#endif  // USE_TDM

__device__ __forceinline__ unsigned short f2h(float f) {
  _Float16 h = (_Float16)f;
  return __builtin_bit_cast(unsigned short, h);
}

// fp32 -> f16 bulk convert (for key_tokens, natural [P][D] layout)
__global__ __launch_bounds__(256) void cvt_f16_kernel(const float* __restrict__ src,
                                                      unsigned short* __restrict__ dst) {
  int i = (blockIdx.x * 256 + threadIdx.x) * 4;
  float4 f = *(const float4*)(src + i);
  unsigned long long pk = (unsigned long long)f2h(f.x)
                        | ((unsigned long long)f2h(f.y) << 16)
                        | ((unsigned long long)f2h(f.z) << 32)
                        | ((unsigned long long)f2h(f.w) << 48);
  *(unsigned long long*)(dst + i) = pk;
}

// fp32 [P][D] -> f16 [D][P] transpose (value_tokens; K-major for WMMA B frags)
__global__ __launch_bounds__(256) void cvt_t_f16_kernel(const float* __restrict__ src,
                                                        unsigned short* __restrict__ dst) {
  int o = blockIdx.x * 256 + threadIdx.x;  // o = d*P + p
  int d = o >> 12;                          // / 4096
  int p = o & (Pp - 1);
  dst[o] = f2h(src[(size_t)p * Dd + d]);
}

__global__ __launch_bounds__(NTHREADS, 1)
void pattention_fused_kernel(const float* __restrict__ inp,          // [B*T][D] f32
                             const unsigned short* __restrict__ Kh,  // [P][D]  f16
                             const unsigned short* __restrict__ Vt,  // [D][P]  f16
                             float* __restrict__ out) {               // [B*T][D] f32
  extern __shared__ unsigned char smem[];
  unsigned short* Qs = (unsigned short*)(smem);            // [BM][Dd]  64 KB
  unsigned short* Ks = (unsigned short*)(smem + 65536);    // [PC][Dd]  64 KB
  unsigned short* Vs = (unsigned short*)(smem + 131072);   // [Dd][PC]  64 KB
  unsigned short* Ss = (unsigned short*)(smem + 196608);   // [BM][PC]   8 KB
  float* part        = (float*)(smem + 204800);            // [4][BM]
  float* rscale      = (float*)(smem + 205824);            // [BM]

  const int tid  = threadIdx.x;
  const int w    = tid >> 5;
  const int ln   = tid & 31;
  const int half = ln >> 4;   // K-half selector for 16-bit A/B fragments
  const int lidx = ln & 15;   // M row (A) / N col (B, C/D)
  const int mi   = w >> 2;    // 0..3 : M subtile (shared by both GEMMs)
  const int pj   = w & 3;     // 0..3 : GEMM1 P-subtile, GEMM2 D column-group
  const int m0   = blockIdx.x * BM;

  // Per the CDNA5 ISA 16-bit fragment layouts:
  //   A (16x32): VGPR v holds K pair at (v>>2)*16 + half*8 + (v&3)*2
  //   B (32x16): VGPR v holds K pair at (v>>2)*16 + half*8 + (v&3)*2
  int kA[8], kB[8];
  #pragma unroll
  for (int v = 0; v < 8; ++v) {
    kA[v] = (v >> 2) * 16 + half * 8 + (v & 3) * 2;
    kB[v] = (v >> 2) * 16 + half * 8 + (v & 3) * 2;
  }

  // Stage Q tile once: fp32 -> f16 into LDS [BM][Dd]
  {
    const float* Qg = inp + (size_t)m0 * Dd;
    for (int i = tid; i < BM * Dd / 4; i += NTHREADS) {
      float4 f = ((const float4*)Qg)[i];
      unsigned long long pk = (unsigned long long)f2h(f.x)
                            | ((unsigned long long)f2h(f.y) << 16)
                            | ((unsigned long long)f2h(f.z) << 32)
                            | ((unsigned long long)f2h(f.w) << 48);
      *(unsigned long long*)(Qs + i * 4) = pk;
    }
  }

  const v8f vzero = {0.f, 0.f, 0.f, 0.f, 0.f, 0.f, 0.f, 0.f};
  v8f acc[8];
  #pragma unroll
  for (int t = 0; t < 8; ++t) acc[t] = vzero;
  float ssq[8] = {0.f, 0.f, 0.f, 0.f, 0.f, 0.f, 0.f, 0.f};

  for (int p0 = 0; p0 < Pp; p0 += PC) {
    __syncthreads();  // previous GEMM2 consumers done before restaging

#ifdef USE_TDM
    // One wave drives the Tensor Data Mover for both chunk tiles, waits on
    // TENSORcnt, then releases the block at the barrier.
    if (w == 0) {
      // K chunk: PC rows x Dd f16, contiguous rows -> Ks[p][d]
      tdm_load_tile_2d(Kh + (size_t)p0 * Dd, Ks, Dd, PC, Dd);
      // V chunk: Dd rows x PC f16, global row stride P -> Vs[d][p]
      tdm_load_tile_2d(Vt + p0, Vs, PC, Dd, Pp);
      __builtin_amdgcn_s_wait_tensorcnt(0);
    }
#else
    // Stage K chunk: Kh[p0 .. p0+PC) x [Dd] -> Ks (straight copy, f16)
    {
      const uint4* kg = (const uint4*)(Kh + (size_t)p0 * Dd);
      uint4* ks = (uint4*)Ks;
      for (int i = tid; i < PC * Dd / 8; i += NTHREADS) ks[i] = kg[i];
    }
    // Stage V chunk: Vt[d][p0 .. p0+PC) -> Vs [Dd][PC]
    {
      uint4* vs = (uint4*)Vs;
      for (int i = tid; i < Dd * PC / 8; i += NTHREADS) {
        int d = i >> 3, u = i & 7;
        vs[i] = ((const uint4*)(Vt + (size_t)d * Pp + p0))[u];
      }
    }
#endif
    // Prefetch next chunk into caches (global_prefetch_b8)
    if (p0 + PC < Pp) {
      __builtin_prefetch(Kh + (size_t)(p0 + PC) * Dd + (size_t)tid * 64, 0, 1);
      __builtin_prefetch(Vt + (size_t)tid * Pp + (p0 + PC), 0, 1);
    }
    __syncthreads();

    // ---- GEMM1: S[64x64] = Q[64x512] x K^T ; wave (mi,pj) owns one 16x16 tile
    v8f c = vzero;
    const unsigned short* Arow = Qs + (size_t)(mi * 16 + lidx) * Dd;
    const unsigned short* Brow = Ks + (size_t)(pj * 16 + lidx) * Dd;
    #pragma unroll 4
    for (int k0 = 0; k0 < Dd; k0 += 32) {
      FragU a, b;
      #pragma unroll
      for (int v = 0; v < 8; ++v) a.u[v] = *(const unsigned int*)(Arow + k0 + kA[v]);
      #pragma unroll
      for (int v = 0; v < 8; ++v) b.u[v] = *(const unsigned int*)(Brow + k0 + kB[v]);
      c = __builtin_amdgcn_wmma_f32_16x16x32_f16(false, a.h, false, b.h,
                                                 (short)0, c, false, false);
    }

    // Exact GELU, accumulate row sum-of-squares, stash S as f16 for GEMM2
    #pragma unroll
    for (int r = 0; r < 8; ++r) {
      float x = c[r];
      float g = 0.5f * x * (1.0f + erff(x * 0.70710678118654752f));
      ssq[r] += g * g;
      Ss[(size_t)(mi * 16 + r + 8 * half) * PC + pj * 16 + lidx] = f2h(g);
    }
    __syncthreads();

    // ---- GEMM2: O[64x512] += S[64x64] x V[64x512]; wave (mi,pj) owns 16x128
    const unsigned short* A2row = Ss + (size_t)(mi * 16 + lidx) * PC;
    #pragma unroll
    for (int kk = 0; kk < PC; kk += 32) {
      FragU a2;
      #pragma unroll
      for (int v = 0; v < 8; ++v) a2.u[v] = *(const unsigned int*)(A2row + kk + kA[v]);
      #pragma unroll
      for (int t = 0; t < 8; ++t) {
        const unsigned short* B2row =
            Vs + (size_t)((pj * 8 + t) * 16 + lidx) * PC;  // Vs is [d][p]
        FragU b2;
        #pragma unroll
        for (int v = 0; v < 8; ++v) b2.u[v] = *(const unsigned int*)(B2row + kk + kB[v]);
        acc[t] = __builtin_amdgcn_wmma_f32_16x16x32_f16(false, a2.h, false, b2.h,
                                                        (short)0, acc[t], false, false);
      }
    }
  }

  // Deterministic row sum-of-squares reduction: butterfly within each 16-lane
  // half (rows differ per half), then fixed-order 4-way sum across pj waves.
  #pragma unroll
  for (int r = 0; r < 8; ++r) {
    float s = ssq[r];
    s += __shfl_xor(s, 1, 32);
    s += __shfl_xor(s, 2, 32);
    s += __shfl_xor(s, 4, 32);
    s += __shfl_xor(s, 8, 32);
    ssq[r] = s;
  }
  if (lidx == 0) {
    #pragma unroll
    for (int r = 0; r < 8; ++r)
      part[pj * BM + (mi * 16 + r + 8 * half)] = ssq[r];
  }
  __syncthreads();
  if (tid < BM) {
    float s = part[0 * BM + tid] + part[1 * BM + tid]
            + part[2 * BM + tid] + part[3 * BM + tid];
    rscale[tid] = 64.0f / sqrtf(s);   // sqrt(P) = 64
  }
  __syncthreads();

  // Scale + store output tile (f32)
  float* op = out + (size_t)m0 * Dd;
  #pragma unroll
  for (int t = 0; t < 8; ++t) {
    #pragma unroll
    for (int r = 0; r < 8; ++r) {
      int m_l = mi * 16 + r + 8 * half;
      int d   = (pj * 8 + t) * 16 + lidx;
      op[(size_t)m_l * Dd + d] = acc[t][r] * rscale[m_l];
    }
  }
}

extern "C" void kernel_launch(void* const* d_in, const int* in_sizes, int n_in,
                              void* d_out, int out_size, void* d_ws, size_t ws_size,
                              hipStream_t stream) {
  (void)in_sizes; (void)n_in; (void)out_size; (void)ws_size;
  const float* inp = (const float*)d_in[0];
  const float* key = (const float*)d_in[1];
  const float* val = (const float*)d_in[2];
  float* out = (float*)d_out;

  // Workspace: f16 key_tokens [P][D] then f16 transposed value_tokens [D][P]
  unsigned short* Kh = (unsigned short*)d_ws;
  unsigned short* Vt = Kh + (size_t)Pp * Dd;

  cvt_f16_kernel<<<(Pp * Dd) / (256 * 4), 256, 0, stream>>>(key, Kh);
  cvt_t_f16_kernel<<<(Pp * Dd) / 256, 256, 0, stream>>>(val, Vt);

  constexpr int SMEM_BYTES = 206080;  // Qs+Ks+Vs+Ss+part+rscale
  pattention_fused_kernel<<<(Bb * Tt) / BM, NTHREADS, SMEM_BYTES, stream>>>(
      inp, Kh, Vt, out);
}
